// PinSAGEModel_28965259444451
// MI455X (gfx1250) — matrix operand order repaired and checked by reading, not compiled, not run on hardware
//
#include <hip/hip_runtime.h>

typedef __attribute__((ext_vector_type(16))) __bf16 v16bf;
typedef __attribute__((ext_vector_type(8)))  float  v8f;

// ---------------- WMMA helpers (bf16 in, f32 accum) ----------------
// A-fragment: 16x32 bf16 tile, row-major in LDS with leading dim lda.
// Lane L: m = L&15, grp = L>>4; elems 0..7 = K=kbase+grp*8+i, elems 8..15 = K=kbase+16+grp*8+i.
__device__ inline v16bf load_a_frag(const __bf16* A, int lda, int kbase, int lane) {
  int m = lane & 15, grp = lane >> 4;
  const __bf16* row = A + m * lda + kbase + grp * 8;
  v16bf a;
#pragma unroll
  for (int i = 0; i < 8; ++i) { a[i] = row[i]; a[i + 8] = row[16 + i]; }
  return a;
}
// B-fragment from pre-swizzled weights: lane's 16 bf16 are contiguous (32B aligned).
// Layout: Bp[((kt*8 + nt)*32 + lane)*16 + i]
__device__ inline v16bf load_b_frag_sw(const __bf16* Bp, int kt, int nt, int lane) {
  return *(const v16bf*)(Bp + (((kt * 8 + nt) * 32 + lane) << 4));
}
__device__ inline v8f wmma_bf16(v16bf a, v16bf b, v8f c) {
  return __builtin_amdgcn_wmma_f32_16x16x32_bf16(false, a, false, b, (short)0, c, false, false);
}

// ---------------- weight f32 -> bf16, fragment-swizzled, zero K-padding ----------------
// dst[((kt*8+nt)*32+lane)*16+i] = src[(kt*32+(lane>>4)*8+(i&7)+(i>>3)*16)*128 + nt*16+(lane&15)]
__global__ void swizzle_w_kernel(const float* __restrict__ src, __bf16* __restrict__ dst,
                                 int rows, int KT) {
  int idx = blockIdx.x * 256 + threadIdx.x;
  int total = KT * 8 * 512;
  if (idx >= total) return;
  int i    = idx & 15;
  int lane = (idx >> 4) & 31;
  int frag = idx >> 9;              // kt*8 + nt
  int kt = frag >> 3, nt = frag & 7;
  int r = kt * 32 + (lane >> 4) * 8 + (i & 7) + ((i >> 3) << 4);
  int c = nt * 16 + (lane & 15);
  dst[idx] = (r < rows) ? (__bf16)src[r * 128 + c] : (__bf16)0.0f;
}

// ---------------- proj: h = feats @ fcW + fcb + track_emb[ids] ----------------
// feats = [cat_embs gather (144) | genre (20) | zero pad -> 192]; 32 rows/block, 8 waves.
__global__ __launch_bounds__(256) void proj_kernel(
    const int* __restrict__ ids, const int* __restrict__ cats,
    const float* __restrict__ genre, const float* __restrict__ track_emb,
    const float* __restrict__ cat_embs, const __bf16* __restrict__ fcW,
    const float* __restrict__ fcb, float* __restrict__ h) {
  __shared__ __bf16 A[32 * 192];
  int tid = threadIdx.x;
  int base_row = blockIdx.x * 32;
  for (int idx = tid; idx < 32 * 192; idx += 256) {
    int r = idx / 192, c = idx - r * 192;
    int R = base_row + r;
    float v = 0.0f;
    if (c < 144) {
      int cat = c >> 4, d = c & 15;
      v = cat_embs[(cat * 101 + cats[R * 9 + cat]) * 16 + d];
    } else if (c < 164) {
      v = genre[R * 20 + (c - 144)];
    }
    A[idx] = (__bf16)v;
  }
  __syncthreads();
  int wave = tid >> 5, lane = tid & 31, grp = lane >> 4, nlo = lane & 15;
  int m0 = (wave >> 2) * 16;
  const __bf16* Arow = A + m0 * 192;
#pragma unroll
  for (int t = 0; t < 2; ++t) {
    int nt = (wave & 3) * 2 + t;
    int col = nt * 16 + nlo;
    v8f acc = {};
#pragma unroll
    for (int kt = 0; kt < 6; ++kt)
      acc = wmma_bf16(load_a_frag(Arow, 192, kt * 32, lane),
                      load_b_frag_sw(fcW, kt, nt, lane), acc);
    float fb = fcb[col];
#pragma unroll
    for (int j = 0; j < 8; ++j) {
      long R = base_row + m0 + j + 8 * grp;
      long id = ids[R];
      h[R * 128 + col] = acc[j] + fb + track_emb[id * 128 + col];
    }
  }
}

// ---------------- n = relu(h @ QW + Qb), K=128; 32 rows/block ----------------
__global__ __launch_bounds__(256) void qgemm_relu_kernel(
    const float* __restrict__ hin, const __bf16* __restrict__ QW,
    const float* __restrict__ Qb, float* __restrict__ nout) {
  __shared__ __bf16 A[32 * 128];
  int tid = threadIdx.x;
  long base = (long)blockIdx.x * 32;
  for (int idx = tid; idx < 32 * 128; idx += 256)
    A[idx] = (__bf16)hin[base * 128 + idx];
  __syncthreads();
  int wave = tid >> 5, lane = tid & 31, grp = lane >> 4, nlo = lane & 15;
  int m0 = (wave >> 2) * 16;
  const __bf16* Arow = A + m0 * 128;
#pragma unroll
  for (int t = 0; t < 2; ++t) {
    int nt = (wave & 3) * 2 + t;
    int col = nt * 16 + nlo;
    v8f acc = {};
#pragma unroll
    for (int kt = 0; kt < 4; ++kt)
      acc = wmma_bf16(load_a_frag(Arow, 128, kt * 32, lane),
                      load_b_frag_sw(QW, kt, nt, lane), acc);
    float qb = Qb[col];
#pragma unroll
    for (int j = 0; j < 8; ++j) {
      float v = acc[j] + qb;
      nout[(base + m0 + j + 8 * grp) * 128 + col] = v > 0.0f ? v : 0.0f;
    }
  }
}

// ---------------- weighted segment-sum of messages (one wave per edge) ----------------
__global__ __launch_bounds__(256) void edge_scatter_kernel(
    const float* __restrict__ n, const int* __restrict__ esrc,
    const int* __restrict__ edst, const float* __restrict__ w,
    float* __restrict__ agg, float* __restrict__ wsum, int nE) {
  int e = blockIdx.x * 8 + (threadIdx.x >> 5);
  if (e >= nE) return;
  int lane = threadIdx.x & 31;
  int s = esrc[e], d = edst[e];
  float we = w[e];
  const float* np = n + (long)s * 128 + lane * 4;
  float* ap = agg + (long)d * 128 + lane * 4;
#pragma unroll
  for (int i = 0; i < 4; ++i) atomicAdd(ap + i, np[i] * we);
  if (lane == 0) atomicAdd(wsum + d, we);
}

// ---------------- z = relu([agg/max(ws,1) | hdst] @ WW + Wb); out = z/||z|| (+base) ----------------
__global__ __launch_bounds__(256) void wgemm_norm_kernel(
    const float* __restrict__ agg, const float* __restrict__ wsum,
    const float* __restrict__ hdst, const __bf16* __restrict__ WW,
    const float* __restrict__ Wb, const float* __restrict__ base_add,
    float* __restrict__ out) {
  __shared__ __bf16 A[32 * 256];   // 16 KB
  __shared__ float  Z[32 * 128];   // 16 KB
  __shared__ float  norms[32];
  int tid = threadIdx.x;
  long base = (long)blockIdx.x * 32;
  for (int idx = tid; idx < 32 * 256; idx += 256) {
    int r = idx >> 8, c = idx & 255;
    long R = base + r;
    float v;
    if (c < 128) {
      float sw = wsum[R];
      sw = sw > 1.0f ? sw : 1.0f;
      v = agg[R * 128 + c] / sw;
    } else {
      v = hdst[R * 128 + (c - 128)];
    }
    A[idx] = (__bf16)v;
  }
  __syncthreads();
  int wave = tid >> 5, lane = tid & 31, grp = lane >> 4, nlo = lane & 15;
  int m0 = (wave >> 2) * 16;
  const __bf16* Arow = A + m0 * 256;
#pragma unroll
  for (int t = 0; t < 2; ++t) {
    int nt = (wave & 3) * 2 + t;
    int col = nt * 16 + nlo;
    v8f acc = {};
#pragma unroll
    for (int kt = 0; kt < 8; ++kt)
      acc = wmma_bf16(load_a_frag(Arow, 256, kt * 32, lane),
                      load_b_frag_sw(WW, kt, nt, lane), acc);
    float wb = Wb[col];
#pragma unroll
    for (int j = 0; j < 8; ++j) {
      float v = acc[j] + wb;
      Z[(m0 + j + 8 * grp) * 128 + col] = v > 0.0f ? v : 0.0f;
    }
  }
  __syncthreads();
  if (tid < 32) {
    float ss = 0.0f;
    for (int c = 0; c < 128; ++c) { float z = Z[tid * 128 + c]; ss += z * z; }
    norms[tid] = ss == 0.0f ? 1.0f : sqrtf(ss);
  }
  __syncthreads();
  int col = tid & 127, rhalf = tid >> 7;
  for (int r = 0; r < 16; ++r) {
    int row = rhalf * 16 + r;
    long R = base + row;
    float v = Z[row * 128 + col] / norms[row];
    if (base_add) v += base_add[R * 128 + col];
    out[R * 128 + col] = v;
  }
}

// ---------------- scores: one wave per pair; s[0..8191]=pos, s[8192..]=neg ----------------
__global__ __launch_bounds__(256) void score_kernel(
    const float* __restrict__ hitem, const int* __restrict__ psrc,
    const int* __restrict__ pdst, const int* __restrict__ nsrc,
    const int* __restrict__ ndst, const int* __restrict__ nids,
    const float* __restrict__ bias, float* __restrict__ s) {
  int p = blockIdx.x * 8 + (threadIdx.x >> 5);
  if (p >= 16384) return;
  int lane = threadIdx.x & 31;
  int src, dst;
  if (p < 8192) { src = psrc[p]; dst = pdst[p]; }
  else          { src = nsrc[p - 8192]; dst = ndst[p - 8192]; }
  const float* a = hitem + (long)src * 128 + lane * 4;
  const float* b = hitem + (long)dst * 128 + lane * 4;
  float acc = 0.0f;
#pragma unroll
  for (int i = 0; i < 4; ++i) acc += a[i] * b[i];
#pragma unroll
  for (int off = 16; off > 0; off >>= 1) acc += __shfl_xor(acc, off, 32);
  if (lane == 0) s[p] = acc + bias[nids[src]] + bias[nids[dst]];
}

__global__ void loss_kernel(const float* __restrict__ s, float* __restrict__ out) {
  int i = blockIdx.x * 256 + threadIdx.x;
  if (i < 8192) {
    float v = s[8192 + i] - s[i] + 1.0f;
    out[i] = v > 0.0f ? v : 0.0f;
  }
}

// rank_i = #{j: s_j < s_i} + #{j<i: s_j == s_i} (stable argsort); sum ranks of positives
__global__ __launch_bounds__(256) void auc_rank_kernel(const float* __restrict__ s,
                                                       double* __restrict__ accum) {
  int i = blockIdx.x * 8 + (threadIdx.x >> 5);
  if (i >= 16384) return;
  int lane = threadIdx.x & 31;
  float vi = s[i];
  int cnt = 0;
  for (int j = lane; j < 16384; j += 32) {
    float vj = s[j];
    cnt += (vj < vi) || (vj == vi && j < i);
  }
#pragma unroll
  for (int off = 16; off > 0; off >>= 1) cnt += __shfl_xor(cnt, off, 32);
  if (lane == 0 && i < 8192) atomicAdd(accum, (double)cnt);
}

__global__ void auc_final_kernel(const double* __restrict__ accum, float* __restrict__ out) {
  double npos = 8192.0, nneg = 8192.0;
  out[8192] = (float)((*accum - npos * (npos - 1.0) * 0.5) / (npos * nneg));
}

// ---------------- driver ----------------
extern "C" void kernel_launch(void* const* d_in, const int* in_sizes, int n_in,
                              void* d_out, int out_size, void* d_ws, size_t ws_size,
                              hipStream_t stream) {
  const int*   src0_id   = (const int*)d_in[0];
  const int*   src0_cats = (const int*)d_in[1];
  const float* src0_gnr  = (const float*)d_in[2];
  const int*   esrc0     = (const int*)d_in[3];
  const int*   edst0     = (const int*)d_in[4];
  const float* w0        = (const float*)d_in[5];
  const int*   esrc1     = (const int*)d_in[6];
  const int*   edst1     = (const int*)d_in[7];
  const float* w1        = (const float*)d_in[8];
  const int*   pos_src   = (const int*)d_in[9];
  const int*   pos_dst   = (const int*)d_in[10];
  const int*   neg_src   = (const int*)d_in[11];
  const int*   neg_dst   = (const int*)d_in[12];
  const int*   seed_nids = (const int*)d_in[13];
  const float* track_emb = (const float*)d_in[14];
  const float* cat_embs  = (const float*)d_in[15];
  const float* fc_W = (const float*)d_in[16];
  const float* fc_b = (const float*)d_in[17];
  const float* Q0_W = (const float*)d_in[18];
  const float* Q0_b = (const float*)d_in[19];
  const float* W0_W = (const float*)d_in[20];
  const float* W0_b = (const float*)d_in[21];
  const float* Q1_W = (const float*)d_in[22];
  const float* Q1_b = (const float*)d_in[23];
  const float* W1_W = (const float*)d_in[24];
  const float* W1_b = (const float*)d_in[25];
  const float* bias = (const float*)d_in[26];

  float* ws = (float*)d_ws;
  float* h     = ws;                            // 262144*128
  float* n0    = ws + 33554432ll;               // 262144*128 (region reused below)
  float* h1    = n0;                            // 65536*128 (n0 dead by then)
  float* n1    = n0 + 8388608ll;                // 65536*128
  float* agg0  = ws + 67108864ll;               // 65536*128
  float* wsum0 = ws + 75497472ll;               // 65536
  float* agg1  = ws + 75563008ll;               // 8192*128
  float* wsum1 = ws + 76611584ll;               // 8192
  float* hitem = ws + 76619776ll;               // 8192*128
  float* s     = ws + 77668352ll;               // 16384
  double* auc_acc = (double*)(ws + 77684736ll); // 8B aligned
  __bf16* fcWb = (__bf16*)(ws + 77684744ll);    // 32B aligned
  __bf16* Q0b = fcWb + 24576;  // fc: KT=6 -> 24576
  __bf16* W0b = Q0b  + 16384;  // Q:  KT=4 -> 16384
  __bf16* Q1b = W0b  + 32768;  // W:  KT=8 -> 32768
  __bf16* W1b = Q1b  + 16384;

  swizzle_w_kernel<<<(6 * 8 * 512 + 255) / 256, 256, 0, stream>>>(fc_W, fcWb, 164, 6);
  swizzle_w_kernel<<<(4 * 8 * 512 + 255) / 256, 256, 0, stream>>>(Q0_W, Q0b, 128, 4);
  swizzle_w_kernel<<<(8 * 8 * 512 + 255) / 256, 256, 0, stream>>>(W0_W, W0b, 256, 8);
  swizzle_w_kernel<<<(4 * 8 * 512 + 255) / 256, 256, 0, stream>>>(Q1_W, Q1b, 128, 4);
  swizzle_w_kernel<<<(8 * 8 * 512 + 255) / 256, 256, 0, stream>>>(W1_W, W1b, 256, 8);

  hipMemsetAsync(agg0, 0, (size_t)8388608 * 4, stream);
  hipMemsetAsync(wsum0, 0, (size_t)65536 * 4, stream);
  hipMemsetAsync(agg1, 0, (size_t)1048576 * 4, stream);
  hipMemsetAsync(wsum1, 0, (size_t)8192 * 4, stream);
  hipMemsetAsync(auc_acc, 0, 8, stream);

  proj_kernel<<<262144 / 32, 256, 0, stream>>>(src0_id, src0_cats, src0_gnr, track_emb,
                                               cat_embs, fcWb, fc_b, h);
  qgemm_relu_kernel<<<262144 / 32, 256, 0, stream>>>(h, Q0b, Q0_b, n0);
  edge_scatter_kernel<<<1048576 / 8, 256, 0, stream>>>(n0, esrc0, edst0, w0, agg0, wsum0, 1048576);
  wgemm_norm_kernel<<<65536 / 32, 256, 0, stream>>>(agg0, wsum0, h, W0b, W0_b, nullptr, h1);
  qgemm_relu_kernel<<<65536 / 32, 256, 0, stream>>>(h1, Q1b, Q1_b, n1);
  edge_scatter_kernel<<<131072 / 8, 256, 0, stream>>>(n1, esrc1, edst1, w1, agg1, wsum1, 131072);
  wgemm_norm_kernel<<<8192 / 32, 256, 0, stream>>>(agg1, wsum1, h1, W1b, W1_b, h, hitem);
  score_kernel<<<16384 / 8, 256, 0, stream>>>(hitem, pos_src, pos_dst, neg_src, neg_dst,
                                              seed_nids, bias, s);
  loss_kernel<<<8192 / 256, 256, 0, stream>>>(s, (float*)d_out);
  auc_rank_kernel<<<16384 / 8, 256, 0, stream>>>(s, auc_acc);
  auc_final_kernel<<<1, 1, 0, stream>>>(auc_acc, (float*)d_out);
}
